// attention_9775345565755
// MI455X (gfx1250) — matrix operand order repaired and checked by reading, not compiled, not run on hardware
//
#include <hip/hip_runtime.h>
#include <math.h>

#define H_DIM 1024
#define N_HEADS 16
#define HEAD_DIM 64

typedef __bf16 v16bf __attribute__((ext_vector_type(16)));
typedef float  v8f   __attribute__((ext_vector_type(8)));

// ---------- bf16 helpers (bit-exact RNE conversion) ----------
static __device__ __forceinline__ __bf16 f2bf(float x) {
    union { float f; unsigned int u; } a; a.f = x;
    unsigned int r = (a.u + 0x7FFFu + ((a.u >> 16) & 1u)) >> 16;
    union { unsigned short s; __bf16 b; } o; o.s = (unsigned short)r;
    return o.b;
}
static __device__ __forceinline__ float bf2f(__bf16 x) {
    union { __bf16 b; unsigned short s; } i; i.b = x;
    union { unsigned int u; float f; } o; o.u = ((unsigned int)i.s) << 16;
    return o.f;
}
static __device__ __forceinline__ unsigned int pack2bf(float lo, float hi) {
    union { __bf16 b[2]; unsigned int u; } p;
    p.b[0] = f2bf(lo); p.b[1] = f2bf(hi);
    return p.u;
}

// ---------- WMMA fragment loaders (CDNA5 16-bit layouts, CDNA5 ISA §7.12.2) ----------
// A 16x32 (MxK): lane L -> m = L%16 ; half h: K = (h/8)*16 + (L/16)*8 + (h%8)
static __device__ __forceinline__ v16bf load_frag_a(const __bf16* base, int ld, int k0) {
    int lane = threadIdx.x & 31;
    int m = lane & 15, hi = lane >> 4;
    const __bf16* p = base + (size_t)m * ld + k0 + hi * 8;
    union { v16bf v; uint4 q[2]; } u;
    u.q[0] = *(const uint4*)p;
    u.q[1] = *(const uint4*)(p + 16);
    return u.v;
}
// B 32x16 (KxN), B[k][n] = rows[n][k]: lane L -> n = L%16 ; half h: k = (h/8)*16 + (L/16)*8 + (h%8)
static __device__ __forceinline__ v16bf load_frag_b(const __bf16* base, int ld, int k0) {
    int lane = threadIdx.x & 31;
    int n = lane & 15, hi = lane >> 4;
    const __bf16* p = base + (size_t)n * ld + k0 + hi * 8;
    union { v16bf v; uint4 q[2]; } u;
    u.q[0] = *(const uint4*)p;
    u.q[1] = *(const uint4*)(p + 16);
    return u.v;
}

// ---------- DPP-based 16-lane reductions (VALU only; no LDS pipe) ----------
template <int CTRL>
static __device__ __forceinline__ float dpp_permf(float v) {
    int r = __builtin_amdgcn_update_dpp(0, __builtin_bit_cast(int, v), CTRL, 0xF, 0xF, true);
    return __builtin_bit_cast(float, r);
}
static __device__ __forceinline__ float redmax16(float v) {
    v = fmaxf(v, dpp_permf<0xB1>(v));    // quad_perm xor1
    v = fmaxf(v, dpp_permf<0x4E>(v));    // quad_perm xor2
    v = fmaxf(v, dpp_permf<0x141>(v));   // row_half_mirror
    v = fmaxf(v, dpp_permf<0x140>(v));   // row_mirror
    return v;
}
static __device__ __forceinline__ float redsum16(float v) {
    v += dpp_permf<0xB1>(v);
    v += dpp_permf<0x4E>(v);
    v += dpp_permf<0x141>(v);
    v += dpp_permf<0x140>(v);
    return v;
}

static __device__ __forceinline__ void store_out(float* p, float v)  { *p = v; }
static __device__ __forceinline__ void store_out(__bf16* p, float v) { *p = f2bf(v); }

// ---------- fp32 -> bf16 conversion, vectorized ----------
__global__ void cvt_f32_bf16(const float4* __restrict__ src, uint2* __restrict__ dst, size_t n4) {
    size_t i = (size_t)blockIdx.x * blockDim.x + threadIdx.x;
    size_t stride = (size_t)gridDim.x * blockDim.x;
    for (; i < n4; i += stride) {
        float4 f = src[i];
        uint2 o;
        o.x = pack2bf(f.x, f.y);
        o.y = pack2bf(f.z, f.w);
        dst[i] = o;
    }
}

// ---------- GEMM: C[s][o] = sum_k A[s][k]*W[o][k]  (y = X @ W^T) ----------
// Register-blocked: each wave computes a 32(M) x 128(N) tile -> 16 WMMAs per 20 b128 loads.
template <typename OutT>
__global__ __launch_bounds__(256) void gemm_bf16_wmma(
    const __bf16* __restrict__ A, const __bf16* __restrict__ W,
    OutT* __restrict__ C, int S, int N, int K) {
    int wave = threadIdx.x >> 5;
    int lane = threadIdx.x & 31;
    int row0 = (blockIdx.y * 8 + wave) * 32;
    int col0 = blockIdx.x * 128;
    if (row0 >= S) return;

    v8f acc[2][8] = {};
    const __bf16* A0 = A + (size_t)row0 * K;
    const __bf16* A1 = A + (size_t)(row0 + 16) * K;

#pragma unroll 2
    for (int k = 0; k < K; k += 32) {
        v16bf a0 = load_frag_a(A0, K, k);
        v16bf a1 = load_frag_a(A1, K, k);
#pragma unroll
        for (int j = 0; j < 8; ++j) {
            v16bf b = load_frag_b(W + (size_t)(col0 + j * 16) * K, K, k);
            acc[0][j] = __builtin_amdgcn_wmma_f32_16x16x32_bf16(false, a0, false, b, (short)0, acc[0][j], false, false);
            acc[1][j] = __builtin_amdgcn_wmma_f32_16x16x32_bf16(false, a1, false, b, (short)0, acc[1][j], false, false);
        }
    }

    int n = lane & 15, hi = lane >> 4;
#pragma unroll
    for (int mt = 0; mt < 2; ++mt)
#pragma unroll
        for (int j = 0; j < 8; ++j)
#pragma unroll
            for (int r = 0; r < 8; ++r) {
                int row = row0 + mt * 16 + r + 8 * hi;  // C layout: VGPR r holds row r+8*(L/16)
                store_out(&C[(size_t)row * N + col0 + j * 16 + n], acc[mt][j][r]);
            }
}

// ---------- RoPE in place on bf16 Q,K; pair (d, d+32) shares inv_freq/cos/sin ----------
__global__ void rope_inplace(__bf16* __restrict__ Q, __bf16* __restrict__ Kk, int S, float base) {
    int idx = blockIdx.x * blockDim.x + threadIdx.x;
    if (idx >= S * N_HEADS) return;
    int s = idx / N_HEADS, head = idx % N_HEADS;
    __bf16* q = Q  + (size_t)s * H_DIM + head * HEAD_DIM;
    __bf16* k = Kk + (size_t)s * H_DIM + head * HEAD_DIM;
    float qv[HEAD_DIM], kv[HEAD_DIM];
#pragma unroll
    for (int d = 0; d < HEAD_DIM; ++d) { qv[d] = bf2f(q[d]); kv[d] = bf2f(k[d]); }
#pragma unroll 4
    for (int i = 0; i < 32; ++i) {
        float invf = powf(base, -((float)(2 * i)) / (float)HEAD_DIM);
        float f = (float)s * invf;
        float c = cosf(f), sn = sinf(f);
        q[i]      = f2bf(qv[i]      * c - qv[2 * i + 1] * sn);
        k[i]      = f2bf(kv[i]      * c - kv[2 * i + 1] * sn);
        q[i + 32] = f2bf(qv[i + 32] * c + qv[2 * i]     * sn);
        k[i + 32] = f2bf(kv[i + 32] * c + kv[2 * i]     * sn);
    }
}

// ---------- Flash attention: per (head, 128-row q block), 8 waves x 16 q-rows ----------
// Double-buffered LDS staging: ONE barrier per 32-wide k-tile; global loads for tile i+1
// are issued before the compute on tile i so the VMEM latency hides under 8 WMMAs + softmax.
__global__ __launch_bounds__(256) void flash_attn_wmma(
    const __bf16* __restrict__ Q, const __bf16* __restrict__ Km,
    const __bf16* __restrict__ V, __bf16* __restrict__ O, int S) {
    __shared__ __bf16 Ks[2][32 * 64];    // K tile, row-major (t, d)
    __shared__ __bf16 Vt[2][64 * 32];    // V tile TRANSPOSED (d, t) -> b128 fragment reads
    __shared__ __bf16 Ps[8][16 * 32];    // per-wave P scratch
    const int H = H_DIM;
    int wave = threadIdx.x >> 5, lane = threadIdx.x & 31;
    int head = blockIdx.y;
    int qbase = blockIdx.x * 128 + wave * 16;
    int n = lane & 15, hi = lane >> 4;

    const __bf16* Qh = Q + (size_t)qbase * H + head * HEAD_DIM;
    v16bf qf0 = load_frag_a(Qh, H, 0);
    v16bf qf1 = load_frag_a(Qh, H, 32);

    v8f acc0 = {}, acc1 = {}, acc2 = {}, acc3 = {};
    float m_i[8], l_i[8];
#pragma unroll
    for (int r = 0; r < 8; ++r) { m_i[r] = -1e30f; l_i[r] = 0.f; }

    // staging split over 256 threads: K: 1 uint4/thread; V: 2 rows x 4 cols/thread
    int kr = threadIdx.x >> 3, kc = (threadIdx.x & 7) * 8;
    int vr = (threadIdx.x >> 4) * 2, vc = (threadIdx.x & 15) * 4;
    __bf16* Pw = &Ps[wave][0];
    const float score_scale = 8.0f;  // reference: attn = QK^T / (d^-0.5) == * sqrt(64)

    uint4 kreg;
    uint2 vreg0, vreg1;
    auto gload = [&](int t0) {
        kreg  = *(const uint4*)(Km + (size_t)(t0 + kr) * H + head * HEAD_DIM + kc);
        const __bf16* vg = V + (size_t)(t0 + vr) * H + head * HEAD_DIM + vc;
        vreg0 = *(const uint2*)vg;
        vreg1 = *(const uint2*)(vg + H);
    };
    auto sstore = [&](int buf) {
        *(uint4*)&Ks[buf][kr * 64 + kc] = kreg;
        union { uint2 q; __bf16 b[4]; } r0, r1;
        r0.q = vreg0; r1.q = vreg1;
#pragma unroll
        for (int h = 0; h < 4; ++h) {  // Vt[d][t], two t's packed per b32 store
            union { __bf16 b[2]; unsigned int u; } p;
            p.b[0] = r0.b[h]; p.b[1] = r1.b[h];
            *(unsigned int*)&Vt[buf][(vc + h) * 32 + vr] = p.u;
        }
    };

    gload(0);
    sstore(0);
    int cur = 0;
    for (int t0 = 0; t0 < S; t0 += 32) {
        __syncthreads();  // publishes buf[cur]; also: everyone is done reading buf[cur^1]
        int t0n = t0 + 32;
        if (t0n < S) gload(t0n);  // VMEM for next tile in flight under this tile's math

        // scores: 16(q) x 32(t), contraction over head_dim (2 WMMA steps per t-half)
        const __bf16* Kc = &Ks[cur][0];
        v8f s0 = {}, s1 = {};
        s0 = __builtin_amdgcn_wmma_f32_16x16x32_bf16(false, qf0, false, load_frag_b(Kc, 64, 0),  (short)0, s0, false, false);
        s0 = __builtin_amdgcn_wmma_f32_16x16x32_bf16(false, qf1, false, load_frag_b(Kc, 64, 32), (short)0, s0, false, false);
        s1 = __builtin_amdgcn_wmma_f32_16x16x32_bf16(false, qf0, false, load_frag_b(Kc + 16 * 64, 64, 0),  (short)0, s1, false, false);
        s1 = __builtin_amdgcn_wmma_f32_16x16x32_bf16(false, qf1, false, load_frag_b(Kc + 16 * 64, 64, 32), (short)0, s1, false, false);

        // online softmax; row r+8*hi lives in VGPR r, spread across 16 lanes (DPP reductions)
#pragma unroll
        for (int r = 0; r < 8; ++r) {
            float a0 = s0[r] * score_scale, a1 = s1[r] * score_scale;
            float mx = redmax16(fmaxf(a0, a1));
            float mnew = fmaxf(m_i[r], mx);
            float alpha = __expf(m_i[r] - mnew);
            m_i[r] = mnew;
            a0 = __expf(a0 - mnew);
            a1 = __expf(a1 - mnew);
            l_i[r] = l_i[r] * alpha + redsum16(a0 + a1);
            acc0[r] *= alpha; acc1[r] *= alpha; acc2[r] *= alpha; acc3[r] *= alpha;
            Pw[(r + 8 * hi) * 32 + n]      = f2bf(a0);
            Pw[(r + 8 * hi) * 32 + 16 + n] = f2bf(a1);
        }
        asm volatile("" ::: "memory");  // same-wave DS ops are in-order; block compiler reordering

        v16bf pa = load_frag_a(Pw, 32, 0);  // P as A-matrix 16x32 (K-dim = t)
        const __bf16* Vc = &Vt[cur][0];
        acc0 = __builtin_amdgcn_wmma_f32_16x16x32_bf16(false, pa, false, load_frag_b(Vc,           32, 0), (short)0, acc0, false, false);
        acc1 = __builtin_amdgcn_wmma_f32_16x16x32_bf16(false, pa, false, load_frag_b(Vc + 16 * 32, 32, 0), (short)0, acc1, false, false);
        acc2 = __builtin_amdgcn_wmma_f32_16x16x32_bf16(false, pa, false, load_frag_b(Vc + 32 * 32, 32, 0), (short)0, acc2, false, false);
        acc3 = __builtin_amdgcn_wmma_f32_16x16x32_bf16(false, pa, false, load_frag_b(Vc + 48 * 32, 32, 0), (short)0, acc3, false, false);

        if (t0n < S) sstore(cur ^ 1);
        cur ^= 1;
    }

#pragma unroll
    for (int r = 0; r < 8; ++r) {
        int row = qbase + r + 8 * hi;
        float inv = 1.0f / l_i[r];
        __bf16* o = O + (size_t)row * H + head * HEAD_DIM;
        o[n]      = f2bf(acc0[r] * inv);
        o[16 + n] = f2bf(acc1[r] * inv);
        o[32 + n] = f2bf(acc2[r] * inv);
        o[48 + n] = f2bf(acc3[r] * inv);
    }
}

extern "C" void kernel_launch(void* const* d_in, const int* in_sizes, int n_in,
                              void* d_out, int out_size, void* d_ws, size_t ws_size,
                              hipStream_t stream) {
    (void)n_in; (void)out_size; (void)ws_size;
    const float* hs = (const float*)d_in[0];
    const float* Wq = (const float*)d_in[1];
    const float* Wk = (const float*)d_in[2];
    const float* Wv = (const float*)d_in[3];
    const float* Wo = (const float*)d_in[4];

    const int H = H_DIM;
    const int S = in_sizes[0] / H;  // B=1
    const size_t sH = (size_t)S * H;
    const size_t hh = (size_t)H * H;

    // workspace (bf16): X | Wq | Wk | Wv | Wo | Q | K | V | O  (~48 MB for S=4096)
    __bf16* Xb  = (__bf16*)d_ws;
    __bf16* Wqb = Xb  + sH;
    __bf16* Wkb = Wqb + hh;
    __bf16* Wvb = Wkb + hh;
    __bf16* Wob = Wvb + hh;
    __bf16* Qb  = Wob + hh;
    __bf16* Kb  = Qb  + sH;
    __bf16* Vb  = Kb  + sH;
    __bf16* Ob  = Vb  + sH;

    cvt_f32_bf16<<<512, 256, 0, stream>>>((const float4*)hs, (uint2*)Xb,  sH / 4);
    cvt_f32_bf16<<<256, 256, 0, stream>>>((const float4*)Wq, (uint2*)Wqb, hh / 4);
    cvt_f32_bf16<<<256, 256, 0, stream>>>((const float4*)Wk, (uint2*)Wkb, hh / 4);
    cvt_f32_bf16<<<256, 256, 0, stream>>>((const float4*)Wv, (uint2*)Wvb, hh / 4);
    cvt_f32_bf16<<<256, 256, 0, stream>>>((const float4*)Wo, (uint2*)Wob, hh / 4);

    dim3 gg(H / 128, S / 256);  // 8 waves/block, 32x128 tile per wave
    gemm_bf16_wmma<__bf16><<<gg, 256, 0, stream>>>(Xb, Wqb, Qb, S, H, H);
    gemm_bf16_wmma<__bf16><<<gg, 256, 0, stream>>>(Xb, Wkb, Kb, S, H, H);
    gemm_bf16_wmma<__bf16><<<gg, 256, 0, stream>>>(Xb, Wvb, Vb, S, H, H);

    float base = 10000.0f;
    if (S > 2048)  // reference: base scaled when seq_len > MAX_POS (SCALING_FACTOR = 1)
        base = 10000.0f * powf((float)S / 2048.0f, (float)HEAD_DIM / (float)(HEAD_DIM - 2));
    int tot = S * N_HEADS;
    rope_inplace<<<(tot + 255) / 256, 256, 0, stream>>>(Qb, Kb, S, base);

    dim3 ag(S / 128, N_HEADS);
    flash_attn_wmma<<<ag, 256, 0, stream>>>(Qb, Kb, Vb, Ob, S);

    gemm_bf16_wmma<float><<<gg, 256, 0, stream>>>(Ob, Wob, (float*)d_out, S, H, H);
}
